// LLamaAttention_20023137535010
// MI455X (gfx1250) — compile-verified, matrix operand order checked by reading, not run on hardware
//
#include <hip/hip_runtime.h>
#include <hip/hip_bf16.h>

typedef __attribute__((ext_vector_type(16))) _Float16 v16h;
typedef __attribute__((ext_vector_type(8)))  _Float16 h8;
typedef __attribute__((ext_vector_type(8)))  float    v8f;
typedef __attribute__((ext_vector_type(4)))  unsigned int u32x4;
typedef __attribute__((ext_vector_type(4)))  int      i32x4;
typedef __attribute__((ext_vector_type(8)))  int      i32x8;

#define DGLOBAL 4096
#define SEQ     2048
#define NHEAD   32
#define HDIM    128
#define NBATCH  2

// TDM path available? (gfx1250 device pass only; host pass compiles fallback)
#if defined(__gfx1250__) && __has_builtin(__builtin_amdgcn_tensor_load_to_lds) && \
    __has_builtin(__builtin_amdgcn_s_wait_tensorcnt)
#define USE_TDM 1
#else
#define USE_TDM 0
#endif

// ---------------------------------------------------------------------------
// Fragment loader: A/B operand for v_wmma_f32_16x16x32_f16.
// 16-bit A 16x32 layout (ISA 7.12.2): lanes 0-15 hold row M=lane, K = {0..7} in
// VGPR0..3 and {16..23} in VGPR4..7; lanes 16-31 hold the {8..15, 24..31} halves.
// B (32x16) mirrors this with N=lane, so both operands read 2x contiguous
// 8-half chunks from a row-major [row][k] tile.
// ---------------------------------------------------------------------------
__device__ __forceinline__ v16h frag_ld(const _Float16* tile, int ld, int lane) {
    int rr = lane & 15;
    int hs = (lane >> 4) & 1;
    const _Float16* p = tile + rr * ld + hs * 8;
    h8 lo = *(const h8*)(p);
    h8 hi = *(const h8*)(p + 16);
    return __builtin_shufflevector(lo, hi, 0,1,2,3,4,5,6,7,8,9,10,11,12,13,14,15);
}

#if USE_TDM
// ---------------------------------------------------------------------------
// TDM 2D tile load: tile_h rows x 32 halves from a row-major f16 matrix with
// row stride 4096, into LDS with 8-half row padding (LD=40 halves) produced by
// the descriptor's pad fields (interval = 16 DWORDs = 64B row, amount = 4 DWORDs).
// D# packing per CDNA5 ISA §8.3 (group0) / §8.4 (group1).
// ---------------------------------------------------------------------------
__device__ __forceinline__ void tdm_load_tile32(unsigned int lds_addr,
                                                const _Float16* gptr, int tile_h) {
    unsigned long long ga = (unsigned long long)(size_t)gptr;
    u32x4 g0;
    g0[0] = 1u;                                        // count=1, user descriptor
    g0[1] = lds_addr;                                  // lds_addr (bytes)
    g0[2] = (unsigned int)ga;                          // global_addr[31:0]
    g0[3] = ((unsigned int)(ga >> 32) & 0x01FFFFFFu)   // global_addr[56:32]
          | (2u << 30);                                // type = 2 ("image")
    i32x8 g1;
    g1[0] = (1 << 16)                                  // data_size = 2 bytes
          | (1 << 20)                                  // pad_enable
          | (3 << 22)                                  // pad_interval: 16 DWORDs (64B)
          | (3 << 25);                                 // pad_amount: 4 DWORDs (8 halves)
    g1[1] = (DGLOBAL & 0xFFFF) << 16;                  // tensor_dim0[15:0]
    g1[2] = (DGLOBAL & 0xFFFF) << 16;                  // tensor_dim0[31:16]=0 | tensor_dim1[15:0]
    g1[3] = (32 << 16);                                // tensor_dim1[31:16]=0 | tile_dim0=32
    g1[4] = tile_h & 0xFFFF;                           // tile_dim1 | tile_dim2=0
    g1[5] = DGLOBAL;                                   // tensor_dim0_stride[31:0]
    g1[6] = 0;                                         // stride0[47:32] | stride1[15:0]
    g1[7] = 0;
    i32x4 z4 = {0, 0, 0, 0};
#if __clang_major__ >= 23
    i32x8 z8 = {0, 0, 0, 0, 0, 0, 0, 0};
    __builtin_amdgcn_tensor_load_to_lds(g0, g1, z4, z4, z8, 0);
#else
    __builtin_amdgcn_tensor_load_to_lds(g0, g1, z4, z4, 0);
#endif
}
#endif

// ---------------------------------------------------------------------------
// f32 -> f16 elementwise convert (8 elems / thread, 16B vector loads)
// ---------------------------------------------------------------------------
__global__ __launch_bounds__(256) void k_convert(const float* __restrict__ in,
                                                 _Float16* __restrict__ out, int n8) {
    int i = blockIdx.x * blockDim.x + threadIdx.x;
    if (i >= n8) return;
    float4 a = ((const float4*)in)[i * 2 + 0];
    float4 b = ((const float4*)in)[i * 2 + 1];
    h8 o;
    o[0] = (_Float16)a.x; o[1] = (_Float16)a.y; o[2] = (_Float16)a.z; o[3] = (_Float16)a.w;
    o[4] = (_Float16)b.x; o[5] = (_Float16)b.y; o[6] = (_Float16)b.z; o[7] = (_Float16)b.w;
    ((h8*)out)[i] = o;
}

// ---------------------------------------------------------------------------
// f32 -> f16 transposed convert: out[f*4096+d] = in[d*4096+f]
// ---------------------------------------------------------------------------
__global__ __launch_bounds__(256) void k_transpose(const float* __restrict__ in,
                                                   _Float16* __restrict__ out) {
    __shared__ float t[32][33];
    int x = blockIdx.x * 32 + threadIdx.x;
    int ybase = blockIdx.y * 32;
    #pragma unroll
    for (int i = 0; i < 4; ++i)
        t[threadIdx.y + i * 8][threadIdx.x] =
            in[(size_t)(ybase + threadIdx.y + i * 8) * DGLOBAL + x];
    __syncthreads();
    int xo = ybase + threadIdx.x;
    int yob = blockIdx.x * 32;
    #pragma unroll
    for (int i = 0; i < 4; ++i)
        out[(size_t)(yob + threadIdx.y + i * 8) * DGLOBAL + xo] =
            (_Float16)t[threadIdx.x][threadIdx.y + i * 8];
}

// ---------------------------------------------------------------------------
// Tiled WMMA GEMM: C[4096x4096] = A[4096xK] * Bt[NxK]^T
// MODE 0: f16 out + RoPE + 1/sqrt(HD) scale (Q)   MODE 1: f16 out + RoPE (K)
// MODE 2: f16 out (V)                              MODE 3: f32 out (d_out)
// 256 thr (8 waves), 128x128 C-tile, K-step 32, 2x4 subtiles per wave.
// Global->LDS staging via TDM (tensor_load_to_lds) with double buffering:
// wave 0 issues next tile's DMA while all waves run WMMAs on the current one.
// ---------------------------------------------------------------------------
template <int MODE>
__global__ __launch_bounds__(256) void k_gemm(const _Float16* __restrict__ A,
                                              const _Float16* __restrict__ Bt,
                                              void* __restrict__ Out,
                                              const float* __restrict__ cosT,
                                              const float* __restrict__ sinT) {
    constexpr int LD = 40;                 // 80B row stride (16B aligned, skewed)
    __shared__ _Float16 As[2][128 * LD];   // 2 x 10 KB
    __shared__ _Float16 Bs[2][128 * LD];   // 2 x 10 KB
    const int tid  = threadIdx.x;
    const int lane = tid & 31;
    const int wave = tid >> 5;
    const int wm = wave >> 1, wn = wave & 1;   // 4x2 wave grid
    const int m0 = blockIdx.y * 128;
    const int n0 = blockIdx.x * 128;

    v8f acc[2][4] = {};

#if USE_TDM
    const unsigned int ldsA0 = (unsigned int)(size_t)&As[0][0];
    const unsigned int ldsA1 = (unsigned int)(size_t)&As[1][0];
    const unsigned int ldsB0 = (unsigned int)(size_t)&Bs[0][0];
    const unsigned int ldsB1 = (unsigned int)(size_t)&Bs[1][0];
    const _Float16* gA = A  + (size_t)m0 * DGLOBAL;
    const _Float16* gB = Bt + (size_t)n0 * DGLOBAL;

    if (tid < 32) {                       // wave 0 drives the DMA engine
        tdm_load_tile32(ldsA0, gA, 128);
        tdm_load_tile32(ldsB0, gB, 128);
    }
    int cur = 0;
    for (int k0 = 0; k0 < DGLOBAL; k0 += 32) {
        if (tid < 32) __builtin_amdgcn_s_wait_tensorcnt(0);
        __syncthreads();                  // publish buf[cur]; all done reading buf[cur^1]
        if (k0 + 32 < DGLOBAL && tid < 32) {
            tdm_load_tile32(cur ? ldsA0 : ldsA1, gA + k0 + 32, 128);
            tdm_load_tile32(cur ? ldsB0 : ldsB1, gB + k0 + 32, 128);
        }
        v16h af[2], bf[4];
        #pragma unroll
        for (int i = 0; i < 2; ++i)
            af[i] = frag_ld(&As[cur][(wm * 32 + i * 16) * LD], LD, lane);
        #pragma unroll
        for (int j = 0; j < 4; ++j)
            bf[j] = frag_ld(&Bs[cur][(wn * 64 + j * 16) * LD], LD, lane);
        #pragma unroll
        for (int i = 0; i < 2; ++i)
            #pragma unroll
            for (int j = 0; j < 4; ++j)
                acc[i][j] = __builtin_amdgcn_wmma_f32_16x16x32_f16(
                    false, af[i], false, bf[j], (short)0, acc[i][j], false, false);
        cur ^= 1;
    }
#else
    for (int k0 = 0; k0 < DGLOBAL; k0 += 32) {
        __syncthreads();
        #pragma unroll
        for (int i = 0; i < 2; ++i) {     // 512 h8 chunks, 2 per thread
            int c   = tid + i * 256;
            int row = c >> 2;
            int kc  = (c & 3) * 8;
            *(h8*)&As[0][row * LD + kc] = *(const h8*)&A [(size_t)(m0 + row) * DGLOBAL + k0 + kc];
            *(h8*)&Bs[0][row * LD + kc] = *(const h8*)&Bt[(size_t)(n0 + row) * DGLOBAL + k0 + kc];
        }
        __syncthreads();
        v16h af[2], bf[4];
        #pragma unroll
        for (int i = 0; i < 2; ++i)
            af[i] = frag_ld(&As[0][(wm * 32 + i * 16) * LD], LD, lane);
        #pragma unroll
        for (int j = 0; j < 4; ++j)
            bf[j] = frag_ld(&Bs[0][(wn * 64 + j * 16) * LD], LD, lane);
        #pragma unroll
        for (int i = 0; i < 2; ++i)
            #pragma unroll
            for (int j = 0; j < 4; ++j)
                acc[i][j] = __builtin_amdgcn_wmma_f32_16x16x32_f16(
                    false, af[i], false, bf[j], (short)0, acc[i][j], false, false);
    }
#endif

    // Epilogue. 16x16 f32 D layout: lane holds col N=lane%16; VGPR r holds row
    // M=r (lanes 0-15) / r+8 (lanes 16-31). Adjacent columns sit in adjacent
    // lanes, so the RoPE (even,odd) pair is the shfl_xor(.,1) partner.
    const int hs = (lane >> 4) & 1;
    #pragma unroll
    for (int i = 0; i < 2; ++i) {
        #pragma unroll
        for (int j = 0; j < 4; ++j) {
            #pragma unroll
            for (int r = 0; r < 8; ++r) {
                int grow = m0 + wm * 32 + i * 16 + hs * 8 + r;
                int gcol = n0 + wn * 64 + j * 16 + (lane & 15);
                float val = acc[i][j][r];
                if constexpr (MODE == 0 || MODE == 1) {
                    int s  = grow & (SEQ - 1);
                    int pj = (gcol & (HDIM - 1)) >> 1;
                    float c  = cosT[s * (HDIM / 2) + pj];
                    float sn = sinT[s * (HDIM / 2) + pj];
                    float part = __shfl_xor(val, 1);
                    bool even = ((lane & 1) == 0);
                    val = val * c + (even ? -part * sn : part * sn);
                    if constexpr (MODE == 0) val *= 0.08838834764831845f; // 1/sqrt(128)
                }
                if constexpr (MODE == 3)
                    ((float*)Out)[(size_t)grow * DGLOBAL + gcol] = val;
                else
                    ((_Float16*)Out)[(size_t)grow * DGLOBAL + gcol] = (_Float16)val;
            }
        }
    }
}

// ---------------------------------------------------------------------------
// Flash attention: grid (S/64, H, B), 128 thr (4 waves). Wave w owns query
// rows [qm0+16w, +16). Q pre-scaled by 1/sqrt(HD) and RoPE'd. Causal via
// block-diagonal masking; online softmax with row stats replicated across the
// 16-lane halves.
// ---------------------------------------------------------------------------
__global__ __launch_bounds__(128) void k_flash(const _Float16* __restrict__ qh,
                                               const _Float16* __restrict__ kh,
                                               const _Float16* __restrict__ vh,
                                               _Float16* __restrict__ ctx) {
    constexpr int LDK = 136, LDV = 72, LDP = 72;
    __shared__ _Float16 Ks[64 * LDK];   // [key][dim]
    __shared__ _Float16 Vt[128 * LDV];  // [dim][key] (transposed for PV B-operand)
    __shared__ _Float16 Ps[64 * LDP];   // [row][key] P tile (wave-private rows)

    const int tid  = threadIdx.x;
    const int lane = tid & 31;
    const int wave = tid >> 5;
    const int rr = lane & 15, hs = (lane >> 4) & 1;
    const int qm0 = blockIdx.x * 64;
    const int h   = blockIdx.y;
    const int b   = blockIdx.z;

    const _Float16* qbase = qh + ((size_t)(b * SEQ + qm0 + wave * 16)) * DGLOBAL + h * HDIM;
    v16h qa[4];
    #pragma unroll
    for (int kb = 0; kb < 4; ++kb) {
        const _Float16* p = qbase + (size_t)rr * DGLOBAL + kb * 32 + hs * 8;
        h8 lo = *(const h8*)(p);
        h8 hi = *(const h8*)(p + 16);
        qa[kb] = __builtin_shufflevector(lo, hi, 0,1,2,3,4,5,6,7,8,9,10,11,12,13,14,15);
    }

    v8f o[8] = {};
    float rmax[8], rsum[8];
    #pragma unroll
    for (int r = 0; r < 8; ++r) { rmax[r] = -3.0e38f; rsum[r] = 0.0f; }

    const int nblocks = (qm0 >> 6) + 1;
    for (int t = 0; t < nblocks; ++t) {
        const int n0 = t * 64;
        __syncthreads();
        const _Float16* kp = kh + ((size_t)(b * SEQ + n0)) * DGLOBAL + h * HDIM;
        const _Float16* vp = vh + ((size_t)(b * SEQ + n0)) * DGLOBAL + h * HDIM;
        #pragma unroll
        for (int i = 0; i < 8; ++i) {            // 1024 h8 chunks, 8 per thread
            int c   = tid + i * 128;
            int row = c >> 4;
            int kc  = (c & 15) * 8;
            *(h8*)&Ks[row * LDK + kc] = *(const h8*)&kp[(size_t)row * DGLOBAL + kc];
            h8 vv = *(const h8*)&vp[(size_t)row * DGLOBAL + kc];
            #pragma unroll
            for (int j = 0; j < 8; ++j) Vt[(kc + j) * LDV + row] = vv[j];
        }
        __syncthreads();

        // S = Q * K^T (K rows ARE the B-operand columns: transpose is free)
        v8f sacc[4] = {};
        #pragma unroll
        for (int kb = 0; kb < 4; ++kb) {
            #pragma unroll
            for (int j = 0; j < 4; ++j) {
                v16h bf = frag_ld(&Ks[(j * 16) * LDK + kb * 32], LDK, lane);
                sacc[j] = __builtin_amdgcn_wmma_f32_16x16x32_f16(
                    false, qa[kb], false, bf, (short)0, sacc[j], false, false);
            }
        }

        if (n0 == qm0) {                         // diagonal block: causal mask
            #pragma unroll
            for (int j = 0; j < 4; ++j)
                #pragma unroll
                for (int r = 0; r < 8; ++r) {
                    int lr = wave * 16 + hs * 8 + r;
                    int lc = j * 16 + rr;
                    if (lc > lr) sacc[j][r] = -1.0e9f;
                }
        }

        // Online softmax (row stats replicated across each 16-lane half)
        #pragma unroll
        for (int r = 0; r < 8; ++r) {
            float mnew = rmax[r];
            #pragma unroll
            for (int j = 0; j < 4; ++j) mnew = fmaxf(mnew, sacc[j][r]);
            #pragma unroll
            for (int off = 1; off < 16; off <<= 1) mnew = fmaxf(mnew, __shfl_xor(mnew, off));
            float corr = __expf(rmax[r] - mnew);
            rmax[r] = mnew;
            float ps = 0.0f;
            #pragma unroll
            for (int j = 0; j < 4; ++j) {
                float pv = __expf(sacc[j][r] - mnew);
                sacc[j][r] = pv;
                ps += pv;
            }
            #pragma unroll
            for (int off = 1; off < 16; off <<= 1) ps += __shfl_xor(ps, off);
            rsum[r] = rsum[r] * corr + ps;
            #pragma unroll
            for (int n = 0; n < 8; ++n) o[n][r] *= corr;
        }

        // P -> LDS (wave-private rows, no cross-wave barrier needed)
        #pragma unroll
        for (int j = 0; j < 4; ++j)
            #pragma unroll
            for (int r = 0; r < 8; ++r)
                Ps[(wave * 16 + hs * 8 + r) * LDP + j * 16 + rr] = (_Float16)sacc[j][r];

        // O += P * V (V transposed in LDS so B-operand rows are contiguous)
        #pragma unroll
        for (int ks = 0; ks < 2; ++ks) {
            v16h pa = frag_ld(&Ps[(wave * 16) * LDP + ks * 32], LDP, lane);
            #pragma unroll
            for (int n = 0; n < 8; ++n) {
                v16h bf = frag_ld(&Vt[(n * 16) * LDV + ks * 32], LDV, lane);
                o[n] = __builtin_amdgcn_wmma_f32_16x16x32_f16(
                    false, pa, false, bf, (short)0, o[n], false, false);
            }
        }
    }

    float inv[8];
    #pragma unroll
    for (int r = 0; r < 8; ++r) inv[r] = 1.0f / rsum[r];
    #pragma unroll
    for (int n = 0; n < 8; ++n)
        #pragma unroll
        for (int r = 0; r < 8; ++r) {
            size_t row = (size_t)(b * SEQ + qm0 + wave * 16 + hs * 8 + r);
            ctx[row * DGLOBAL + h * HDIM + n * 16 + rr] = (_Float16)(o[n][r] * inv[r]);
        }
}

// ---------------------------------------------------------------------------
extern "C" void kernel_launch(void* const* d_in, const int* in_sizes, int n_in,
                              void* d_out, int out_size, void* d_ws, size_t ws_size,
                              hipStream_t stream) {
    const float* x    = (const float*)d_in[0];
    const float* wq   = (const float*)d_in[1];
    const float* wk   = (const float*)d_in[2];
    const float* wv   = (const float*)d_in[3];
    const float* wo   = (const float*)d_in[4];
    const float* fcos = (const float*)d_in[5];
    const float* fsin = (const float*)d_in[6];
    // d_in[7] = mask (causal handled analytically), d_in[8] = pos_start (== 0, unused by ref)

    const size_t NE = (size_t)DGLOBAL * DGLOBAL;   // 16.7M elements per matrix
    const size_t HB = NE * sizeof(_Float16);       // 32 MB
    char* ws = (char*)d_ws;
    _Float16* xh   = (_Float16*)(ws + 0 * HB);
    _Float16* wqt  = (_Float16*)(ws + 1 * HB);
    _Float16* wkt  = (_Float16*)(ws + 2 * HB);
    _Float16* wvt  = (_Float16*)(ws + 3 * HB);
    _Float16* wot  = (_Float16*)(ws + 4 * HB);
    _Float16* qhd  = (_Float16*)(ws + 5 * HB);
    _Float16* khd  = (_Float16*)(ws + 6 * HB);
    _Float16* vhd  = (_Float16*)(ws + 7 * HB);
    _Float16* ctxh = (_Float16*)(ws + 8 * HB);

    // 1) precision conversion (+ weight transposes)
    {
        int n8 = (int)(NE / 8);
        k_convert<<<dim3((n8 + 255) / 256), dim3(256), 0, stream>>>(x, xh, n8);
        dim3 tg(DGLOBAL / 32, DGLOBAL / 32), tb(32, 8);
        k_transpose<<<tg, tb, 0, stream>>>(wq, wqt);
        k_transpose<<<tg, tb, 0, stream>>>(wk, wkt);
        k_transpose<<<tg, tb, 0, stream>>>(wv, wvt);
        k_transpose<<<tg, tb, 0, stream>>>(wo, wot);
    }

    // 2) QKV projections (RoPE + scale fused into Q/K epilogues)
    dim3 gg(DGLOBAL / 128, DGLOBAL / 128), gb(256);
    k_gemm<0><<<gg, gb, 0, stream>>>(xh, wqt, (void*)qhd, fcos, fsin);
    k_gemm<1><<<gg, gb, 0, stream>>>(xh, wkt, (void*)khd, fcos, fsin);
    k_gemm<2><<<gg, gb, 0, stream>>>(xh, wvt, (void*)vhd, fcos, fsin);

    // 3) flash attention
    k_flash<<<dim3(SEQ / 64, NHEAD, NBATCH), dim3(128), 0, stream>>>(qhd, khd, vhd, ctxh);

    // 4) output projection -> fp32 d_out
    k_gemm<3><<<gg, gb, 0, stream>>>(ctxh, wot, d_out, fcos, fsin);
}
